// GeometricMultiHeadAttention_6030134083837
// MI455X (gfx1250) — matrix operand order repaired and checked by reading, not compiled, not run on hardware
//
#include <hip/hip_runtime.h>
#include <hip/hip_bf16.h>

// ---------------------------------------------------------------------------
// Geometric (IPA-style) multi-head attention for MI455X (gfx1250, wave32).
// All dense contractions go through v_wmma_f32_16x16x32_f16 (f32 accumulate,
// f16 operands converted on the fly).  The 201MB relative-position embedding
// is never materialized: it is recomputed per query row into LDS as f16 tiles
// (laid out so each WMMA B-operand VGPR is one 32-bit ds_load) and consumed
// directly by WMMA (rq.remb and attn.remb kernels).
// ---------------------------------------------------------------------------

#define Lq 512
#define Dm 512
#define Hh 8
#define DH 64
#define PQ 4
#define PV 8
#define ZD 128
#define RQDIM 192   // 3 * 64

typedef __attribute__((ext_vector_type(16))) _Float16 v16h;
typedef __attribute__((ext_vector_type(8)))  float    v8f;
typedef __attribute__((ext_vector_type(2)))  _Float16 h2;

__device__ __forceinline__ float rope_freq(int u) {
    // periods = exp(linspace(log(1), log(100), 32)); freq = 2*pi/period
    const float LOG100 = 4.6051702f;
    return 6.2831853f * __expf(-LOG100 * (float)u * (1.0f / 31.0f));
}

// A-operand K index for lane-half h, vgpr v (contiguous order k = 16h + 2v, matching B)
__device__ __forceinline__ int a_koff(int v, int half) {
    return half * 16 + 2 * v;
}

// ---------------------------------------------------------------------------
// Generic batched WMMA GEMM: C[M,N] (+)= alpha * A[M,K] * B[K,N] (+ bias[N])
// fp32 in/out, f16 WMMA operands.  One wave computes NT adjacent 16x16 tiles
// along N, reusing the A fragment across NT back-to-back WMMAs.
// grid = (N/(16*NT), M/16, batch).  M,N mult of 16; K mult of 32.
// ---------------------------------------------------------------------------
template <int NT, bool ACC, bool BIAS>
__global__ void gemm_wmma_f16(const float* __restrict__ A,
                              const float* __restrict__ B,
                              const float* __restrict__ bias,
                              float* __restrict__ C,
                              int M, int N, int K,
                              long long sA, long long sB, long long sC,
                              float alpha)
{
    const int lane = threadIdx.x & 31;
    const int half = lane >> 4;
    const int mn   = lane & 15;
    const int tn = blockIdx.x, tm = blockIdx.y, bz = blockIdx.z;
    A += (long long)bz * sA;
    B += (long long)bz * sB;
    C += (long long)bz * sC;
    const long long arow = (long long)(tm * 16 + mn) * K;

    v8f acc[NT] = {};
    for (int kk = 0; kk < K; kk += 32) {
        v16h af;
#pragma unroll
        for (int v = 0; v < 8; ++v) {
            // A pairs are K-contiguous -> 64-bit loads
            const float2 a2 = *(const float2*)(A + arow + kk + a_koff(v, half));
            af[2 * v]     = (_Float16)a2.x;
            af[2 * v + 1] = (_Float16)a2.y;
        }
#pragma unroll
        for (int t = 0; t < NT; ++t) {
            const int col = (tn * NT + t) * 16 + mn;
            v16h bf;
#pragma unroll
            for (int v = 0; v < 8; ++v) {
                int kb = kk + half * 16 + 2 * v;
                bf[2 * v]     = (_Float16)B[(long long)kb * N + col];
                bf[2 * v + 1] = (_Float16)B[(long long)(kb + 1) * N + col];
            }
            acc[t] = __builtin_amdgcn_wmma_f32_16x16x32_f16(
                false, af, false, bf, (short)0, acc[t], false, false);
        }
    }
#pragma unroll
    for (int t = 0; t < NT; ++t) {
        const int col = (tn * NT + t) * 16 + mn;
        const float bv = BIAS ? bias[col] : 0.f;
#pragma unroll
        for (int r = 0; r < 8; ++r) {
            int orow = tm * 16 + r + half * 8;
            long long idx = (long long)orow * N + col;
            float vout = acc[t][r] * alpha + bv;
            if (ACC) vout += C[idx];
            C[idx] = vout;
        }
    }
}

// ---------------------------------------------------------------------------
// RoPE expansion: q,k -> 192-dim rotated copies; v -> v_rot (per x component).
// q[i, h*64+c]  ->  qrope[h][i][x*64+c], kropeT[h][x*64+c][i], vrot[h][i][x*64+c]
// ---------------------------------------------------------------------------
__global__ void rope_qkv(const float* __restrict__ q, const float* __restrict__ k,
                         const float* __restrict__ v, const float* __restrict__ trans,
                         float* __restrict__ qrope, float* __restrict__ kropeT,
                         float* __restrict__ vrot)
{
    int idx = blockIdx.x * blockDim.x + threadIdx.x;  // H*L*3*64
    if (idx >= Hh * Lq * 3 * DH) return;
    int c = idx % DH;
    int x = (idx / DH) % 3;
    int i = (idx / (3 * DH)) % Lq;
    int h = idx / (3 * DH * Lq);

    float f = rope_freq(c & 31);
    float ang = trans[i * 3 + x] * f;
    float sn, cs;
    __sincosf(ang, &sn, &cs);

    int src = i * Dm + h * DH;
    float qv = q[src + c];
    float qr = (c < 32) ? -q[src + c + 32] : q[src + c - 32];
    float kv = k[src + c];
    float kr = (c < 32) ? -k[src + c + 32] : k[src + c - 32];
    float vv = v[src + c];
    float vr = (c < 32) ? -v[src + c + 32] : v[src + c - 32];

    int d = x * DH + c;
    qrope[((long long)h * Lq + i) * RQDIM + d] = qv * cs + qr * sn;
    kropeT[((long long)h * RQDIM + d) * Lq + i] = kv * cs + kr * sn;
    vrot[((long long)h * Lq + i) * RQDIM + d] = vv * cs + vr * sn;
}

// ---------------------------------------------------------------------------
// Point features: reorder qpts/kvpts, add translations, squared norms.
// ---------------------------------------------------------------------------
__global__ void points_prep(const float* __restrict__ qraw, const float* __restrict__ kvraw,
                            const float* __restrict__ trans,
                            float* __restrict__ qpR, float* __restrict__ kpR,
                            float* __restrict__ vpR, float* __restrict__ qsq,
                            float* __restrict__ ksq)
{
    int idx = blockIdx.x * blockDim.x + threadIdx.x;  // L*H
    if (idx >= Lq * Hh) return;
    int h = idx % Hh;
    int i = idx / Hh;
    float t0 = trans[i * 3 + 0], t1 = trans[i * 3 + 1], t2 = trans[i * 3 + 2];
    float qs = 0.f, ks = 0.f;
    for (int p = 0; p < PQ; ++p) {
        for (int c = 0; c < 3; ++c) {
            float tc = (c == 0) ? t0 : ((c == 1) ? t1 : t2);
            float qv = qraw[i * (3 * Hh * PQ) + c * (Hh * PQ) + h * PQ + p] + tc;
            qpR[i * (Hh * PQ * 3) + h * (PQ * 3) + p * 3 + c] = qv;
            qs += qv * qv;
        }
    }
    for (int p = 0; p < PQ + PV; ++p) {
        for (int c = 0; c < 3; ++c) {
            float tc = (c == 0) ? t0 : ((c == 1) ? t1 : t2);
            float kv = kvraw[i * (3 * Hh * (PQ + PV)) + c * (Hh * (PQ + PV)) + h * (PQ + PV) + p] + tc;
            if (p < PQ) {
                kpR[i * (Hh * PQ * 3) + h * (PQ * 3) + p * 3 + c] = kv;
                ks += kv * kv;
            } else {
                vpR[i * (Hh * PV * 3) + h * (PV * 3) + (p - PQ) * 3 + c] = kv;
            }
        }
    }
    qsq[i * Hh + h] = qs;
    ksq[i * Hh + h] = ks;
}

// Pad w_b (128x8) to (128x16) for the WMMA GEMM.
__global__ void pad_wb(const float* __restrict__ w_b, float* __restrict__ wbp)
{
    int idx = blockIdx.x * blockDim.x + threadIdx.x;  // ZD*16
    if (idx >= ZD * 16) return;
    int col = idx & 15, row = idx >> 4;
    wbp[idx] = (col < Hh) ? w_b[row * Hh + col] : 0.f;
}

// ---------------------------------------------------------------------------
// attn[h,i,j] += mask + pair-bias + point-distance term (elementwise).
// ---------------------------------------------------------------------------
__global__ void attn_extras(float* __restrict__ attn, const float* __restrict__ zbuf,
                            const unsigned char* __restrict__ mask,
                            const float* __restrict__ qpR, const float* __restrict__ kpR,
                            const float* __restrict__ qsq, const float* __restrict__ ksq,
                            const float* __restrict__ head_w)
{
    long long idx = (long long)blockIdx.x * blockDim.x + threadIdx.x;  // H*L*L
    if (idx >= (long long)Hh * Lq * Lq) return;
    int j = (int)(idx % Lq);
    int i = (int)((idx / Lq) % Lq);
    int h = (int)(idx / ((long long)Lq * Lq));

    float cross = 0.f;
    const float* qp = qpR + i * (Hh * PQ * 3) + h * (PQ * 3);
    const float* kp = kpR + j * (Hh * PQ * 3) + h * (PQ * 3);
#pragma unroll
    for (int t = 0; t < PQ * 3; ++t) cross += qp[t] * kp[t];
    float d2 = qsq[i * Hh + h] + ksq[j * Hh + h] - 2.f * cross;
    float hw = __logf(1.f + __expf(head_w[h])) * 0.23570226f;  // softplus * sqrt(2/36)

    float add = (mask[j] ? 0.f : -1.0e9f)
              + zbuf[((long long)i * Lq + j) * 16 + h]
              - 0.5f * d2 * hw;
    attn[idx] += add;
}

// ---------------------------------------------------------------------------
// attn[h,i,:] += (rq[i,h,:] . remb[i,:,:]) / sqrt(192)
// One block per query row i.  Contraction over d: B[d][j] = remb[i,j,d], so a
// B-operand VGPR holds (d, d+1) for fixed j -> LDS laid out [j][d] so each
// VGPR pair is one aligned 32-bit ds load.  The 6 rq A-fragments are
// chunk-invariant and preloaded once.
// ---------------------------------------------------------------------------
__global__ void __launch_bounds__(128) rq_remb_attn(const float* __restrict__ rq,
                                                    const float* __restrict__ trans,
                                                    float* __restrict__ attn)
{
    __shared__ _Float16 rembJ[128][200];  // [jj][d], d contiguous (51.2KB)
    const int i = blockIdx.x;
    const int tid = threadIdx.x;
    const int lane = tid & 31, wave = tid >> 5;
    const int half = lane >> 4, mn = lane & 15;
    const float ti0 = trans[i * 3 + 0], ti1 = trans[i * 3 + 1], ti2 = trans[i * 3 + 2];

    // Preload the 6 rq A-fragments (rows >= Hh are zero padding).
    v16h afk[RQDIM / 32];
#pragma unroll
    for (int ks = 0; ks < RQDIM / 32; ++ks) {
#pragma unroll
        for (int v = 0; v < 8; ++v) {
            int ka = ks * 32 + a_koff(v, half);
            float a0 = 0.f, a1 = 0.f;
            if (mn < Hh) {
                const float2 a2 = *(const float2*)(rq + i * (Hh * RQDIM) + mn * RQDIM + ka);
                a0 = a2.x; a1 = a2.y;
            }
            afk[ks][2 * v]     = (_Float16)a0;
            afk[ks][2 * v + 1] = (_Float16)a1;
        }
    }

    for (int jc = 0; jc < 4; ++jc) {
        const int j0 = jc * 128;
        for (int e = tid; e < 128 * RQDIM; e += 128) {
            int jj = e / RQDIM, d = e % RQDIM;
            int c = d / DH, u = d % DH;
            float f = rope_freq(u & 31);
            float tic = (c == 0) ? ti0 : ((c == 1) ? ti1 : ti2);
            float ang = (trans[(j0 + jj) * 3 + c] - tic) * f;
            rembJ[jj][d] = (_Float16)((u < 32) ? __cosf(ang) : __sinf(ang));
        }
        __syncthreads();
        for (int t = wave; t < 8; t += 4) {
            const int jl = t * 16 + mn;   // local j column
            v8f acc = {};
#pragma unroll
            for (int ks = 0; ks < RQDIM / 32; ++ks) {
                v16h bf;
#pragma unroll
                for (int v = 0; v < 8; ++v) {
                    int kb = ks * 32 + half * 16 + 2 * v;     // even -> aligned
                    h2 p = *(const h2*)&rembJ[jl][kb];
                    bf[2 * v]     = p.x;
                    bf[2 * v + 1] = p.y;
                }
                acc = __builtin_amdgcn_wmma_f32_16x16x32_f16(
                    false, afk[ks], false, bf, (short)0, acc, false, false);
            }
            const float s = 0.07216878f;  // 1/sqrt(192)
#pragma unroll
            for (int r = 0; r < 8; ++r) {
                int hrow = r + half * 8;
                if (hrow < Hh) {
                    int j = j0 + t * 16 + mn;
                    attn[((long long)hrow * Lq + i) * Lq + j] += acc[r] * s;
                }
            }
        }
        __syncthreads();
    }
}

// ---------------------------------------------------------------------------
// Row softmax over j for each (h,i).
// ---------------------------------------------------------------------------
__global__ void __launch_bounds__(256) softmax_rows(float* __restrict__ attn)
{
    __shared__ float red[256];
    float* p = attn + (long long)blockIdx.x * Lq;
    int tid = threadIdx.x;
    float m = -1e30f;
    for (int j = tid; j < Lq; j += 256) m = fmaxf(m, p[j]);
    red[tid] = m;
    __syncthreads();
    for (int s = 128; s > 0; s >>= 1) {
        if (tid < s) red[tid] = fmaxf(red[tid], red[tid + s]);
        __syncthreads();
    }
    m = red[0];
    __syncthreads();
    float sum = 0.f;
    for (int j = tid; j < Lq; j += 256) {
        float e = __expf(p[j] - m);
        p[j] = e;
        sum += e;
    }
    red[tid] = sum;
    __syncthreads();
    for (int s = 128; s > 0; s >>= 1) {
        if (tid < s) red[tid] += red[tid + s];
        __syncthreads();
    }
    float inv = 1.f / red[0];
    for (int j = tid; j < Lq; j += 256) p[j] *= inv;
}

// ---------------------------------------------------------------------------
// r_out[i, h*192+d] = sum_j attn[h,i,j] * remb[i,j,d]   (K=j contraction)
// One block per i.  B[j][d] = remb[i,j,d]: a B-operand VGPR holds (j, j+1)
// for fixed d -> LDS laid out [d][jj] so each VGPR pair is one aligned
// 32-bit ds load.  A fragments (attn row chunk) preloaded per chunk,
// reused across 3 n-tiles.
// ---------------------------------------------------------------------------
__global__ void __launch_bounds__(128) attn_remb_rout(const float* __restrict__ attn,
                                                      const float* __restrict__ trans,
                                                      float* __restrict__ r_out)
{
    __shared__ _Float16 rembD[RQDIM][136];  // [d][jj], jj contiguous (52.2KB)
    const int i = blockIdx.x;
    const int tid = threadIdx.x;
    const int lane = tid & 31, wave = tid >> 5;
    const int half = lane >> 4, mn = lane & 15;
    const float ti0 = trans[i * 3 + 0], ti1 = trans[i * 3 + 1], ti2 = trans[i * 3 + 2];

    v8f acc[3] = {};
    for (int jc = 0; jc < 4; ++jc) {
        const int j0 = jc * 128;
        for (int e = tid; e < RQDIM * 128; e += 128) {
            int d = e >> 7, jj = e & 127;
            int c = d / DH, u = d % DH;
            float f = rope_freq(u & 31);
            float tic = (c == 0) ? ti0 : ((c == 1) ? ti1 : ti2);
            float ang = (trans[(j0 + jj) * 3 + c] - tic) * f;
            rembD[d][jj] = (_Float16)((u < 32) ? __cosf(ang) : __sinf(ang));
        }
        __syncthreads();

        // Preload the 4 A fragments for this j-chunk (attn rows, zero-padded).
        v16h afk[4];
#pragma unroll
        for (int ks = 0; ks < 4; ++ks) {
#pragma unroll
            for (int v = 0; v < 8; ++v) {
                int ka = j0 + ks * 32 + a_koff(v, half);
                float a0 = 0.f, a1 = 0.f;
                if (mn < Hh) {
                    const float2 a2 = *(const float2*)(attn + ((long long)mn * Lq + i) * Lq + ka);
                    a0 = a2.x; a1 = a2.y;
                }
                afk[ks][2 * v]     = (_Float16)a0;
                afk[ks][2 * v + 1] = (_Float16)a1;
            }
        }
#pragma unroll
        for (int ti = 0; ti < 3; ++ti) {
            int t = wave + ti * 4;        // n-tile 0..11 over the 192 d's
            const int dl = t * 16 + mn;   // d column
#pragma unroll
            for (int ks = 0; ks < 4; ++ks) {
                v16h bf;
#pragma unroll
                for (int v = 0; v < 8; ++v) {
                    int kb = ks * 32 + half * 16 + 2 * v;     // even -> aligned
                    h2 p = *(const h2*)&rembD[dl][kb];
                    bf[2 * v]     = p.x;
                    bf[2 * v + 1] = p.y;
                }
                acc[ti] = __builtin_amdgcn_wmma_f32_16x16x32_f16(
                    false, afk[ks], false, bf, (short)0, acc[ti], false, false);
            }
        }
        __syncthreads();
    }
#pragma unroll
    for (int ti = 0; ti < 3; ++ti) {
        int t = wave + ti * 4;
#pragma unroll
        for (int r = 0; r < 8; ++r) {
            int hrow = r + half * 8;
            if (hrow < Hh)
                r_out[i * (Hh * RQDIM) + hrow * RQDIM + t * 16 + mn] = acc[ti][r];
        }
    }
}

// ---------------------------------------------------------------------------
// o_pt[i,h,p,c] = sum_j attn[h,i,j]*vp[j,h,p,c] - trans[i,c]
// ---------------------------------------------------------------------------
__global__ void opt_kernel(const float* __restrict__ attn, const float* __restrict__ vpR,
                           const float* __restrict__ trans, float* __restrict__ o_pt)
{
    int idx = blockIdx.x * blockDim.x + threadIdx.x;  // L*H*PV*3
    if (idx >= Lq * Hh * PV * 3) return;
    int c = idx % 3;
    int p = (idx / 3) % PV;
    int h = (idx / (3 * PV)) % Hh;
    int i = idx / (3 * PV * Hh);
    const float* arow = attn + ((long long)h * Lq + i) * Lq;
    float s = 0.f;
    for (int j = 0; j < Lq; ++j)
        s += arow[j] * vpR[j * (Hh * PV * 3) + h * (PV * 3) + p * 3 + c];
    o_pt[idx] = s - trans[i * 3 + c];
}

// ipa_feat[i, 0:64]=x, [64:128]=y, [128:192]=z, [192:256]=norm
__global__ void ipa_kernel(const float* __restrict__ o_pt, float* __restrict__ ipa)
{
    int idx = blockIdx.x * blockDim.x + threadIdx.x;  // L * 64
    if (idx >= Lq * Hh * PV) return;
    int hp = idx % (Hh * PV);
    int i = idx / (Hh * PV);
    float x0 = o_pt[i * (Hh * PV * 3) + hp * 3 + 0];
    float x1 = o_pt[i * (Hh * PV * 3) + hp * 3 + 1];
    float x2 = o_pt[i * (Hh * PV * 3) + hp * 3 + 2];
    ipa[i * 256 + hp]        = x0;
    ipa[i * 256 + 64 + hp]   = x1;
    ipa[i * 256 + 128 + hp]  = x2;
    ipa[i * 256 + 192 + hp]  = sqrtf(x0 * x0 + x1 * x1 + x2 * x2 + 1e-6f);
}

// Rotate back by -pos and mean over the 3 spatial copies -> o_flat[i, h*64+c]
__global__ void oflat_kernel(const float* __restrict__ obuf, const float* __restrict__ trans,
                             float* __restrict__ o_flat)
{
    int idx = blockIdx.x * blockDim.x + threadIdx.x;  // L * H * 64
    if (idx >= Lq * Hh * DH) return;
    int c = idx % DH;
    int h = (idx / DH) % Hh;
    int i = idx / (DH * Hh);
    const float* ob = obuf + ((long long)h * Lq + i) * RQDIM;
    float f = rope_freq(c & 31);
    float s = 0.f;
#pragma unroll
    for (int x = 0; x < 3; ++x) {
        float a = trans[i * 3 + x] * f;
        float sn, cs;
        __sincosf(a, &sn, &cs);
        float ov = ob[x * DH + c];
        float orr = (c < 32) ? -ob[x * DH + c + 32] : ob[x * DH + c - 32];
        s += ov * cs - orr * sn;  // rotary with -pos
    }
    o_flat[i * Dm + h * DH + c] = s * (1.f / 3.f);
}

// ---------------------------------------------------------------------------
template <int NT>
static inline void gemm_nt(hipStream_t st, const float* A, const float* B, const float* bias,
                           float* C, int M, int N, int K, long long sA, long long sB,
                           long long sC, int batch, int acc, float alpha)
{
    dim3 g(N / (16 * NT), M / 16, batch);
    dim3 b(32);
    if (acc) {
        if (bias) gemm_wmma_f16<NT, true, true><<<g, b, 0, st>>>(A, B, bias, C, M, N, K, sA, sB, sC, alpha);
        else      gemm_wmma_f16<NT, true, false><<<g, b, 0, st>>>(A, B, bias, C, M, N, K, sA, sB, sC, alpha);
    } else {
        if (bias) gemm_wmma_f16<NT, false, true><<<g, b, 0, st>>>(A, B, bias, C, M, N, K, sA, sB, sC, alpha);
        else      gemm_wmma_f16<NT, false, false><<<g, b, 0, st>>>(A, B, bias, C, M, N, K, sA, sB, sC, alpha);
    }
}

static inline void gemm(hipStream_t st, const float* A, const float* B, const float* bias,
                        float* C, int M, int N, int K, long long sA, long long sB,
                        long long sC, int batch, int acc, float alpha)
{
    if      (N % 64 == 0) gemm_nt<4>(st, A, B, bias, C, M, N, K, sA, sB, sC, batch, acc, alpha);
    else if (N % 32 == 0) gemm_nt<2>(st, A, B, bias, C, M, N, K, sA, sB, sC, batch, acc, alpha);
    else                  gemm_nt<1>(st, A, B, bias, C, M, N, K, sA, sB, sC, batch, acc, alpha);
}

extern "C" void kernel_launch(void* const* d_in, const int* in_sizes, int n_in,
                              void* d_out, int out_size, void* d_ws, size_t ws_size,
                              hipStream_t stream)
{
    const float* x      = (const float*)d_in[0];
    const float* z      = (const float*)d_in[1];
    const unsigned char* mask = (const unsigned char*)d_in[2];
    const float* trans  = (const float*)d_in[3];
    // d_in[4] = rots (unused by reference)
    const float* w_q    = (const float*)d_in[5];
    const float* w_k    = (const float*)d_in[6];
    const float* w_v    = (const float*)d_in[7];
    const float* w_o    = (const float*)d_in[8];
    const float* w_b    = (const float*)d_in[9];
    const float* w_qpts = (const float*)d_in[10];
    const float* b_qpts = (const float*)d_in[11];
    const float* w_kvpts= (const float*)d_in[12];
    const float* b_kvpts= (const float*)d_in[13];
    const float* head_w = (const float*)d_in[14];
    const float* w_pt   = (const float*)d_in[15];
    const float* b_pt   = (const float*)d_in[16];
    const float* w_rq   = (const float*)d_in[17];
    const float* w_r    = (const float*)d_in[18];
    float* out = (float*)d_out;

    float* W = (float*)d_ws;
    size_t off = 0;
    auto alloc = [&](size_t n) { float* p = W + off; off += n; return p; };
    float* q       = alloc((size_t)Lq * Dm);
    float* k       = alloc((size_t)Lq * Dm);
    float* v       = alloc((size_t)Lq * Dm);
    float* qrope   = alloc((size_t)Hh * Lq * RQDIM);
    float* kropeT  = alloc((size_t)Hh * RQDIM * Lq);
    float* vrot    = alloc((size_t)Hh * Lq * RQDIM);
    float* qraw    = alloc((size_t)Lq * Hh * PQ * 3);
    float* kvraw   = alloc((size_t)Lq * Hh * (PQ + PV) * 3);
    float* qpR     = alloc((size_t)Lq * Hh * PQ * 3);
    float* kpR     = alloc((size_t)Lq * Hh * PQ * 3);
    float* vpR     = alloc((size_t)Lq * Hh * PV * 3);
    float* qsq     = alloc((size_t)Lq * Hh);
    float* ksq     = alloc((size_t)Lq * Hh);
    float* wbp     = alloc((size_t)ZD * 16);
    float* zbuf    = alloc((size_t)Lq * Lq * 16);
    float* attn    = alloc((size_t)Hh * Lq * Lq);
    float* rqb     = alloc((size_t)Lq * Hh * RQDIM);
    float* obuf    = alloc((size_t)Hh * Lq * RQDIM);
    float* r_out   = alloc((size_t)Lq * Hh * RQDIM);
    float* o_pt    = alloc((size_t)Lq * Hh * PV * 3);
    float* ipa     = alloc((size_t)Lq * 256);
    float* o_flat  = alloc((size_t)Lq * Dm);
    (void)ws_size; (void)in_sizes; (void)n_in; (void)out_size;

    // --- input projections (WMMA GEMMs) ---
    gemm(stream, x, w_q,    nullptr, q,     Lq, Dm,               Dm, 0, 0, 0, 1, 0, 1.f);
    gemm(stream, x, w_k,    nullptr, k,     Lq, Dm,               Dm, 0, 0, 0, 1, 0, 1.f);
    gemm(stream, x, w_v,    nullptr, v,     Lq, Dm,               Dm, 0, 0, 0, 1, 0, 1.f);
    gemm(stream, x, w_qpts, b_qpts,  qraw,  Lq, Hh * PQ * 3,      Dm, 0, 0, 0, 1, 0, 1.f);
    gemm(stream, x, w_kvpts,b_kvpts, kvraw, Lq, Hh * (PQ+PV) * 3, Dm, 0, 0, 0, 1, 0, 1.f);
    gemm(stream, x, w_rq,   nullptr, rqb,   Lq, Hh * RQDIM,       Dm, 0, 0, 0, 1, 0, 1.f);

    // --- pair bias z @ w_b (w_b padded to N=16) ---
    pad_wb<<<dim3((ZD * 16 + 255) / 256), dim3(256), 0, stream>>>(w_b, wbp);
    gemm(stream, z, wbp, nullptr, zbuf, Lq * Lq, 16, ZD, 0, 0, 0, 1, 0, 1.f);

    // --- rope expansion + point prep ---
    rope_qkv<<<dim3((Hh * Lq * 3 * DH + 255) / 256), dim3(256), 0, stream>>>(
        q, k, v, trans, qrope, kropeT, vrot);
    points_prep<<<dim3((Lq * Hh + 255) / 256), dim3(256), 0, stream>>>(
        qraw, kvraw, trans, qpR, kpR, vpR, qsq, ksq);

    // --- attn = qrope . krope / sqrt(DH)  (batched over heads) ---
    gemm(stream, qrope, kropeT, nullptr, attn, Lq, Lq, RQDIM,
         (long long)Lq * RQDIM, (long long)RQDIM * Lq, (long long)Lq * Lq,
         Hh, 0, 0.125f);

    // --- + mask + pair bias + point-distance terms ---
    attn_extras<<<dim3((Hh * Lq * Lq + 255) / 256), dim3(256), 0, stream>>>(
        attn, zbuf, mask, qpR, kpR, qsq, ksq, head_w);

    // --- + rq . remb / sqrt(192)  (remb recomputed in LDS, WMMA) ---
    rq_remb_attn<<<dim3(Lq), dim3(128), 0, stream>>>(rqb, trans, attn);

    // --- softmax ---
    softmax_rows<<<dim3(Hh * Lq), dim3(256), 0, stream>>>(attn);

    // --- o = attn @ v_rot  (batched over heads) ---
    gemm(stream, attn, vrot, nullptr, obuf, Lq, RQDIM, Lq,
         (long long)Lq * Lq, (long long)Lq * RQDIM, (long long)Lq * RQDIM,
         Hh, 0, 1.f);

    // --- point outputs + ipa features ---
    opt_kernel<<<dim3((Lq * Hh * PV * 3 + 255) / 256), dim3(256), 0, stream>>>(
        attn, vpR, trans, o_pt);
    ipa_kernel<<<dim3((Lq * Hh * PV + 255) / 256), dim3(256), 0, stream>>>(o_pt, ipa);

    // --- r_out = attn . remb  (remb recomputed in LDS, WMMA) ---
    attn_remb_rout<<<dim3(Lq), dim3(128), 0, stream>>>(attn, trans, r_out);

    // --- rotate-back + mean -> o_flat ---
    oflat_kernel<<<dim3((Lq * Hh * DH + 255) / 256), dim3(256), 0, stream>>>(
        obuf, trans, o_flat);

    // --- output projections, accumulated into d_out ---
    gemm(stream, o_flat, w_o,  nullptr, out, Lq, Dm, Dm,          0, 0, 0, 1, 0, 1.f);
    gemm(stream, ipa,    w_pt, b_pt,    out, Lq, Dm, 256,         0, 0, 0, 1, 1, 1.f);
    gemm(stream, r_out,  w_r,  nullptr, out, Lq, Dm, Hh * RQDIM,  0, 0, 0, 1, 1, 1.f);
}
